// Attention_66915590472540
// MI455X (gfx1250) — matrix operand order repaired and checked by reading, not compiled, not run on hardware
//
#include <hip/hip_runtime.h>
#include <hip/hip_bf16.h>

#define SEQ    2048
#define DIM    2048
#define DHEAD  128
#define NHEADS 16
#define BATCH  2
#define TOKENS (BATCH * SEQ)
#define EPSLN  1e-5f
#define MASKV  -1e10f

typedef __attribute__((ext_vector_type(16))) __bf16 v16bf;
typedef __attribute__((ext_vector_type(8)))  float  v8f;

union FragAB {
    v16bf v;
    unsigned int u[8];
    unsigned short h[16];
};

__device__ __forceinline__ unsigned short f2bf(float f) {
    unsigned int u = __float_as_uint(f);
    u += 0x7fffu + ((u >> 16) & 1u);   // round-to-nearest-even
    return (unsigned short)(u >> 16);
}
__device__ __forceinline__ float bf2f(unsigned short h) {
    return __uint_as_float(((unsigned int)h) << 16);
}

__device__ __forceinline__ v8f wmma_bf16(v16bf a, v16bf b, v8f c) {
    return __builtin_amdgcn_wmma_f32_16x16x32_bf16(
        /*neg_a=*/false, a, /*neg_b=*/false, b,
        /*c_mod=*/(short)0, c, /*reuse_a=*/false, /*reuse_b=*/false);
}

__device__ __forceinline__ float redmax16(float v) {
    v = fmaxf(v, __shfl_xor(v, 1, 16));
    v = fmaxf(v, __shfl_xor(v, 2, 16));
    v = fmaxf(v, __shfl_xor(v, 4, 16));
    v = fmaxf(v, __shfl_xor(v, 8, 16));
    return v;
}
__device__ __forceinline__ float redsum16(float v) {
    v += __shfl_xor(v, 1, 16);
    v += __shfl_xor(v, 2, 16);
    v += __shfl_xor(v, 4, 16);
    v += __shfl_xor(v, 8, 16);
    return v;
}

// ---------------- LayerNorm: one block per token row, fp32 in, bf16 out ----
__global__ void __launch_bounds__(256)
ln_kernel(const float* __restrict__ x,
          const float* __restrict__ gamma,
          unsigned short* __restrict__ xn) {
    const int row = blockIdx.x;
    const float4* x4 = (const float4*)(x + (size_t)row * DIM);
    __shared__ float s_sum[8], s_sq[8];
    float sum = 0.f, sq = 0.f;
    float4 vals[2];
    for (int i = 0; i < 2; ++i) {
        float4 v = x4[threadIdx.x + i * 256];
        vals[i] = v;
        sum += v.x + v.y + v.z + v.w;
        sq  += v.x * v.x + v.y * v.y + v.z * v.z + v.w * v.w;
    }
    for (int m = 1; m < 32; m <<= 1) {
        sum += __shfl_xor(sum, m, 32);
        sq  += __shfl_xor(sq,  m, 32);
    }
    const int wid = threadIdx.x >> 5;
    if ((threadIdx.x & 31) == 0) { s_sum[wid] = sum; s_sq[wid] = sq; }
    __syncthreads();
    if (threadIdx.x == 0) {
        float ts = 0.f, tq = 0.f;
        for (int i = 0; i < 8; ++i) { ts += s_sum[i]; tq += s_sq[i]; }
        s_sum[0] = ts; s_sq[0] = tq;
    }
    __syncthreads();
    const float mean = s_sum[0] * (1.f / DIM);
    const float var  = s_sq[0] * (1.f / DIM) - mean * mean;
    const float rs   = rsqrtf(var + EPSLN);
    unsigned short* xo = xn + (size_t)row * DIM;
    for (int i = 0; i < 2; ++i) {
        const int base = (threadIdx.x + i * 256) * 4;
        float4 v = vals[i];
        xo[base + 0] = f2bf(rs * (v.x - mean) * gamma[base + 0]);
        xo[base + 1] = f2bf(rs * (v.y - mean) * gamma[base + 1]);
        xo[base + 2] = f2bf(rs * (v.z - mean) * gamma[base + 2]);
        xo[base + 3] = f2bf(rs * (v.w - mean) * gamma[base + 3]);
    }
}

// ---------------- GEMM: C[M,N] = A_bf16[M,K] @ B_f32[K,N] ------------------
// Block tile 128x64, 8 waves each computing 32x32 via 2x2 WMMA 16x16x32 tiles.
// B is converted to bf16 and stored transposed in LDS so B-fragment reads are
// contiguous dword ds_loads matching the 32x16 bf16 B-operand layout.
template <bool OUT_F32>
__global__ void __launch_bounds__(256)
gemm_bf16(const unsigned short* __restrict__ A,
          const float* __restrict__ B,
          void* __restrict__ C,
          int M, int N, int K) {
    __shared__ unsigned short sA[128][36];   // [m][k]
    __shared__ unsigned short sBT[64][36];   // [n][k]  (transposed)
    const int tid  = threadIdx.x;
    const int lane = tid & 31;
    const int wave = tid >> 5;
    const int wm = (wave >> 1) * 32;
    const int wn = (wave & 1) * 32;
    const int m0 = blockIdx.y * 128;
    const int n0 = blockIdx.x * 64;
    const unsigned int* Au = (const unsigned int*)A;
    const int K2 = K >> 1;

    v8f acc[2][2];
    for (int mi = 0; mi < 2; ++mi)
        for (int ni = 0; ni < 2; ++ni)
            for (int rr = 0; rr < 8; ++rr) acc[mi][ni][rr] = 0.f;

    for (int k0 = 0; k0 < K; k0 += 32) {
        // stage A tile (128x32 bf16) via dword loads
        for (int i = 0; i < 8; ++i) {
            int u = tid + i * 256;          // 0..2047 dwords
            int r = u >> 4, cp = u & 15;
            *(unsigned int*)&sA[r][cp * 2] =
                Au[(size_t)(m0 + r) * K2 + (k0 >> 1) + cp];
        }
        // stage B tile (32x64 f32 -> bf16, transposed)
        for (int i = 0; i < 8; ++i) {
            int u = tid + i * 256;          // 0..2047 elems
            int kk = u >> 6, nn = u & 63;
            sBT[nn][kk] = f2bf(B[(size_t)(k0 + kk) * N + n0 + nn]);
        }
        // prefetch next K tile into cache (global_prefetch_b8)
        if (k0 + 32 < K) {
            int u = tid * 8;
            int r = u >> 4;
            __builtin_prefetch(&Au[(size_t)(m0 + r) * K2 + ((k0 + 32) >> 1)], 0, 3);
            int kk = tid >> 3;
            if (kk < 32)
                __builtin_prefetch(&B[(size_t)(k0 + 32 + kk) * N + n0], 0, 3);
        }
        __syncthreads();

        FragAB af[2], bfr[2];
        const int r   = lane & 15;
        const int kbA = (lane < 16) ? 0 : 16;
        const int kbB = (lane < 16) ? 0 : 16;
        for (int mi = 0; mi < 2; ++mi) {
            const unsigned short* rowp = &sA[wm + mi * 16 + r][0];
            for (int i = 0; i < 4; ++i) {
                af[mi].u[i]     = *(const unsigned int*)&rowp[kbA + 2 * i];
                af[mi].u[4 + i] = *(const unsigned int*)&rowp[kbA + 8 + 2 * i];
            }
        }
        for (int ni = 0; ni < 2; ++ni) {
            const unsigned short* colp = &sBT[wn + ni * 16 + r][0];
            for (int i = 0; i < 8; ++i)
                bfr[ni].u[i] = *(const unsigned int*)&colp[kbB + 2 * i];
        }
        for (int mi = 0; mi < 2; ++mi)
            for (int ni = 0; ni < 2; ++ni)
                acc[mi][ni] = wmma_bf16(af[mi].v, bfr[ni].v, acc[mi][ni]);
        __syncthreads();
    }

    const int cn   = lane & 15;
    const int rofs = (lane < 16) ? 0 : 8;
    for (int mi = 0; mi < 2; ++mi)
        for (int ni = 0; ni < 2; ++ni)
            for (int rr = 0; rr < 8; ++rr) {
                int row = m0 + wm + mi * 16 + rr + rofs;
                int col = n0 + wn + ni * 16 + cn;
                float val = acc[mi][ni][rr];
                if constexpr (OUT_F32)
                    ((float*)C)[(size_t)row * N + col] = val;
                else
                    ((unsigned short*)C)[(size_t)row * N + col] = f2bf(val);
            }
}

// ---------------- Rotary (in-place on bf16, pair-wise) ---------------------
__global__ void rotary_kernel(unsigned short* __restrict__ buf,
                              const float* __restrict__ sinp,
                              const float* __restrict__ cosp,
                              int rowlen, float scale) {
    const int ppr = rowlen >> 1;
    const int total = TOKENS * ppr;
    int idx = blockIdx.x * blockDim.x + threadIdx.x;
    if (idx >= total) return;
    int row = idx / ppr;
    int pr  = idx - row * ppr;
    int pos = row & (SEQ - 1);
    int dh  = (pr * 2) & (DHEAD - 1);
    float s  = sinp[pos * DHEAD + dh];
    float co = cosp[pos * DHEAD + dh];
    unsigned int* p = (unsigned int*)buf + (size_t)row * ppr + pr;
    unsigned int d = *p;
    float x0 = bf2f((unsigned short)(d & 0xffff));
    float x1 = bf2f((unsigned short)(d >> 16));
    float o0 = (x0 * co - x1 * s) * scale;
    float o1 = (x1 * co + x0 * s) * scale;
    *p = (unsigned int)f2bf(o0) | ((unsigned int)f2bf(o1) << 16);
}

// ---------------- Flash attention: one wave per 16-query tile --------------
// __launch_bounds__(32, 1): single-wave workgroup, occupancy-1 OK -> let the
// compiler keep Q fragments, O accumulators and softmax state in VGPRs
// (wave32 can allocate up to 1024 VGPRs; spilling to scratch is far worse).
__global__ void __launch_bounds__(32, 1)
fa_kernel(const unsigned short* __restrict__ qbf,  // [T, H*D]
          const unsigned short* __restrict__ kbf,  // [T, D]
          const unsigned short* __restrict__ vbf,  // [T, D]
          unsigned short* __restrict__ obf) {      // [T, H*D]
    __shared__ unsigned short sP[16][36];
    const int lane  = threadIdx.x;
    const int qt    = blockIdx.x;
    const int head  = blockIdx.y;
    const int batch = blockIdx.z;
    const int qpos0 = qt * 16;
    const int qtok0 = batch * SEQ + qpos0;
    const unsigned int* qu = (const unsigned int*)qbf;
    const unsigned int* ku = (const unsigned int*)kbf;
    const int r16  = lane & 15;
    const int half = (lane < 16) ? 0 : 1;
    const int kbA  = half * 16;
    const int kbB  = half * 16;

    // Q A-fragments: 4 chunks of K=32 over DHEAD=128
    FragAB aq[4];
    {
        const size_t rowbase =
            (size_t)(qtok0 + r16) * (DIM / 2) + head * (DHEAD / 2);
        for (int ch = 0; ch < 4; ++ch)
            for (int i = 0; i < 4; ++i) {
                aq[ch].u[i]     = qu[rowbase + ch * 16 + (kbA >> 1) + i];
                aq[ch].u[4 + i] = qu[rowbase + ch * 16 + 4 + (kbA >> 1) + i];
            }
    }

    v8f oacc[8];
    float mrow[8], srow[8];
    for (int rj = 0; rj < 8; ++rj) {
        mrow[rj] = -1e30f;
        srow[rj] = 0.f;
        for (int dt = 0; dt < 8; ++dt) oacc[dt][rj] = 0.f;
    }

    const int nkb = ((qpos0 + 15) >> 5) + 1;   // 32-key causal blocks
    for (int kb = 0; kb < nkb; ++kb) {
        const int j0 = kb * 32;
        // prefetch next key/value block while this one computes
        if (kb + 1 < nkb) {
            __builtin_prefetch(
                &kbf[(size_t)(batch * SEQ + j0 + 32 + lane) * DHEAD], 0, 3);
            __builtin_prefetch(
                &vbf[(size_t)(batch * SEQ + j0 + 32 + lane) * DHEAD], 0, 3);
        }
        // S = Q K^T for 32 keys (two 16x16 C tiles)
        v8f s0, s1;
        for (int rj = 0; rj < 8; ++rj) { s0[rj] = 0.f; s1[rj] = 0.f; }
        for (int ch = 0; ch < 4; ++ch) {
            FragAB b0, b1;
            for (int i = 0; i < 8; ++i) {
                const int dof = ch * 16 + (kbB >> 1) + i;     // dword offset in d
                b0.u[i] = ku[(size_t)(batch * SEQ + j0 + r16) * (DHEAD / 2) + dof];
                b1.u[i] = ku[(size_t)(batch * SEQ + j0 + 16 + r16) * (DHEAD / 2) + dof];
            }
            s0 = wmma_bf16(aq[ch].v, b0.v, s0);
            s1 = wmma_bf16(aq[ch].v, b1.v, s1);
        }
        // causal mask
        for (int rj = 0; rj < 8; ++rj) {
            const int qpos = qpos0 + rj + half * 8;
            if (j0 + r16 > qpos)      s0[rj] = MASKV;
            if (j0 + 16 + r16 > qpos) s1[rj] = MASKV;
        }
        // online softmax
        float p0[8], p1[8];
        for (int rj = 0; rj < 8; ++rj) {
            float tm = redmax16(fmaxf(s0[rj], s1[rj]));
            float mn = fmaxf(mrow[rj], tm);
            float alpha = __expf(mrow[rj] - mn);
            mrow[rj] = mn;
            p0[rj] = __expf(s0[rj] - mn);
            p1[rj] = __expf(s1[rj] - mn);
            float rs = redsum16(p0[rj] + p1[rj]);
            srow[rj] = srow[rj] * alpha + rs;
            for (int dt = 0; dt < 8; ++dt) oacc[dt][rj] *= alpha;
        }
        // bounce P through LDS to re-layout C-tile -> A-fragment (16x32 bf16)
        __syncthreads();
        for (int rj = 0; rj < 8; ++rj) {
            sP[rj + half * 8][r16]      = f2bf(p0[rj]);
            sP[rj + half * 8][16 + r16] = f2bf(p1[rj]);
        }
        __syncthreads();
        FragAB ap;
        for (int i = 0; i < 4; ++i) {
            ap.u[i]     = *(const unsigned int*)&sP[r16][kbA + 2 * i];
            ap.u[4 + i] = *(const unsigned int*)&sP[r16][kbA + 8 + 2 * i];
        }
        // O += P @ V  (8 d-tiles of 16)
        for (int dt = 0; dt < 8; ++dt) {
            FragAB bv;
            for (int hh = 0; hh < 16; ++hh) {
                const int kk = kbB + hh;
                bv.h[hh] = vbf[(size_t)(batch * SEQ + j0 + kk) * DHEAD +
                               dt * 16 + r16];
            }
            oacc[dt] = wmma_bf16(ap.v, bv.v, oacc[dt]);
        }
    }

    for (int dt = 0; dt < 8; ++dt)
        for (int rj = 0; rj < 8; ++rj) {
            const int row = qtok0 + rj + half * 8;
            const int col = head * DHEAD + dt * 16 + r16;
            obf[(size_t)row * DIM + col] = f2bf(oacc[dt][rj] / srow[rj]);
        }
}

// ---------------------------------------------------------------------------
extern "C" void kernel_launch(void* const* d_in, const int* in_sizes, int n_in,
                              void* d_out, int out_size, void* d_ws,
                              size_t ws_size, hipStream_t stream) {
    (void)in_sizes; (void)n_in; (void)out_size; (void)ws_size;
    const float* x     = (const float*)d_in[0];
    const float* gamma = (const float*)d_in[1];
    const float* wq    = (const float*)d_in[2];
    const float* wk    = (const float*)d_in[3];
    const float* wv    = (const float*)d_in[4];
    const float* wo    = (const float*)d_in[5];
    const float* sinp  = (const float*)d_in[6];
    const float* cosp  = (const float*)d_in[7];
    // d_in[8] causal_mask: causality computed analytically in fa_kernel

    char* ws = (char*)d_ws;
    size_t off = 0;
    auto carve = [&](size_t bytes) -> char* {
        char* p = ws + off;
        off += (bytes + 255) & ~(size_t)255;
        return p;
    };
    unsigned short* xn   = (unsigned short*)carve((size_t)TOKENS * DIM * 2);
    unsigned short* qb   = (unsigned short*)carve((size_t)TOKENS * DIM * 2);
    unsigned short* kbuf = (unsigned short*)carve((size_t)TOKENS * DHEAD * 2);
    unsigned short* vbuf = (unsigned short*)carve((size_t)TOKENS * DHEAD * 2);
    unsigned short* ob   = (unsigned short*)carve((size_t)TOKENS * DIM * 2);

    ln_kernel<<<TOKENS, 256, 0, stream>>>(x, gamma, xn);

    gemm_bf16<false><<<dim3(DIM / 64, TOKENS / 128), 256, 0, stream>>>(
        xn, wq, qb, TOKENS, DIM, DIM);
    gemm_bf16<false><<<dim3(DHEAD / 64, TOKENS / 128), 256, 0, stream>>>(
        xn, wk, kbuf, TOKENS, DHEAD, DIM);
    gemm_bf16<false><<<dim3(DHEAD / 64, TOKENS / 128), 256, 0, stream>>>(
        xn, wv, vbuf, TOKENS, DHEAD, DIM);

    const float qscale = 0.08838834764831845f;  // 128^-0.5
    {
        int qpairs = TOKENS * (DIM / 2);
        rotary_kernel<<<(qpairs + 255) / 256, 256, 0, stream>>>(
            qb, sinp, cosp, DIM, qscale);
        int kpairs = TOKENS * (DHEAD / 2);
        rotary_kernel<<<(kpairs + 255) / 256, 256, 0, stream>>>(
            kbuf, sinp, cosp, DHEAD, 1.0f);
    }

    fa_kernel<<<dim3(SEQ / 16, NHEADS, BATCH), 32, 0, stream>>>(
        qb, kbuf, vbuf, ob);

    gemm_bf16<true><<<dim3(DIM / 64, TOKENS / 128), 256, 0, stream>>>(
        ob, wo, (float*)d_out, TOKENS, DIM, DIM);
}
